// CoconutGPT_Same_Word_Embedding_15187004358728
// MI455X (gfx1250) — matrix operand order repaired and checked by reading, not compiled, not run on hardware
//
#include <hip/hip_runtime.h>
#include <hip/hip_bf16.h>
#include <math.h>

// ---------------- problem constants (from reference setup_inputs) -----------
constexpr int B_  = 4;
constexpr int S_  = 512;
constexpr int V_  = 50257;
constexpr int D_  = 768;
constexpr int H_  = 12;
constexpr int NL_ = 2;
constexpr int DFF_ = 3072;
constexpr int DH_ = 64;          // D/H
constexpr float SCALE_ = 0.125f; // 1/sqrt(64)
constexpr int LATENT_START_ = 64;
constexpr int N_LATENT_ = 6;

typedef unsigned short ushort;
typedef __attribute__((ext_vector_type(16))) __bf16 v16bf;
typedef __attribute__((ext_vector_type(8)))  float  v8f;

union FragBF {
    uint4 q[2];
    v16bf v;
};

__device__ __forceinline__ ushort f2bf(float f) {
    unsigned u = __float_as_uint(f);
    unsigned r = (u + 0x7FFFu + ((u >> 16) & 1u)) >> 16;   // RNE
    return (ushort)r;
}

// ---------------- conversion kernels ---------------------------------------
__global__ __launch_bounds__(256) void k_convert(const float* __restrict__ in,
                                                 ushort* __restrict__ out, long n) {
    long i = (long)blockIdx.x * blockDim.x + threadIdx.x;
    if (i < n) out[i] = f2bf(in[i]);
}

// in: [L][R][C] f32  ->  out: [L][C][R] bf16  (transpose + convert)
__global__ __launch_bounds__(256) void k_convert_t(const float* __restrict__ in,
                                                   ushort* __restrict__ out,
                                                   int L, int R, int C) {
    long n = (long)L * R * C;
    long i = (long)blockIdx.x * blockDim.x + threadIdx.x;
    if (i >= n) return;
    int l = (int)(i / ((long)R * C));
    long rem = i - (long)l * R * C;
    int r = (int)(rem / C);
    int c = (int)(rem - (long)r * C);
    out[(long)l * R * C + (long)c * R + r] = f2bf(in[i]);
}

// embeds[b,s,:] = wte[input_ids[b,s], :]
__global__ __launch_bounds__(256) void k_embed(const int* __restrict__ ids,
                                               const float* __restrict__ wte,
                                               float* __restrict__ embeds) {
    long n = (long)B_ * S_ * D_;
    long i = (long)blockIdx.x * blockDim.x + threadIdx.x;
    if (i >= n) return;
    long row = i / D_;
    int  d   = (int)(i - row * D_);
    embeds[i] = wte[(long)ids[row] * D_ + d];
}

// x[b,t,:] = embeds[b,t,:] + wpe[t,:]   for t < T
__global__ __launch_bounds__(256) void k_xinit(const float* __restrict__ embeds,
                                               const float* __restrict__ wpe,
                                               float* __restrict__ x, int T) {
    long n = (long)B_ * T * D_;
    long i = (long)blockIdx.x * blockDim.x + threadIdx.x;
    if (i >= n) return;
    int b = (int)(i / ((long)T * D_));
    long rem = i - (long)b * T * D_;
    int t = (int)(rem / D_);
    int d = (int)(rem - (long)t * D_);
    x[i] = embeds[((long)b * S_ + t) * D_ + d] + wpe[(long)t * D_ + d];
}

// ---------------- layernorm (one wave32 per row of D=768) ------------------
__global__ __launch_bounds__(256) void k_ln(const float* __restrict__ x,
                                            const float* __restrict__ g,
                                            const float* __restrict__ bta,
                                            ushort* __restrict__ ob,
                                            float* of, int M) {
    int lane = threadIdx.x & 31;
    int w    = threadIdx.x >> 5;
    long row = (long)blockIdx.x * 8 + w;
    if (row >= M) return;
    const float* xr = x + row * D_;
    float s = 0.f;
    for (int i = lane; i < D_; i += 32) s += xr[i];
    for (int o = 16; o >= 1; o >>= 1) s += __shfl_xor(s, o, 32);
    float mean = s * (1.0f / D_);
    float v = 0.f;
    for (int i = lane; i < D_; i += 32) { float d = xr[i] - mean; v += d * d; }
    for (int o = 16; o >= 1; o >>= 1) v += __shfl_xor(v, o, 32);
    float rstd = rsqrtf(v * (1.0f / D_) + 1e-5f);
    for (int i = lane; i < D_; i += 32) {
        float y = (xr[i] - mean) * rstd * g[i] + bta[i];
        ob[row * D_ + i] = f2bf(y);
        if (of) of[row * D_ + i] = y;
    }
}

// ---------------- WMMA bf16 GEMM, 64x64 macro-tile per wave ----------------
// C = A[M,K](bf16,row-major) @ Wt[N,K]^T (bf16, weight pre-transposed [N,K])
// flags: 1 = +bias[n], 2 = gelu(tanh), 4 = +resid[cidx], 8 = bf16 output (Cb)
// Row mapping: m -> batch = m / rpb, r = m % rpb
//   A row   : A + batch*aBatch + r*K
//   C index : batch*cBatch + r*N + n
__global__ __launch_bounds__(256) void k_gemm(
    const ushort* __restrict__ A, const ushort* __restrict__ Wt,
    const float* __restrict__ bias, const float* resid,
    float* Cf, ushort* Cb,
    int M, int N, int K, int rpb,
    long long aBatch, long long cBatch, int flags) {

    const int lane = threadIdx.x & 31;
    const int wave = threadIdx.x >> 5;
    const int nGn  = (N + 63) >> 6;               // 64-wide N groups
    const int nGm  = (M + 63) >> 6;               // 64-tall M groups
    long long tile = (long long)blockIdx.x * (blockDim.x >> 5) + wave;
    if (tile >= (long long)nGm * nGn) return;
    const int mg = (int)(tile / nGn);             // adjacent waves share A via cache
    const int ng = (int)(tile % nGn);

    const int hl = lane >> 4;                     // K-half selector
    const int mr = lane & 15;

    // A row pointers for the 4 M-frags (rows clamped; stores are guarded)
    const ushort* Arow[4];
#pragma unroll
    for (int i = 0; i < 4; ++i) {
        int mA = mg * 64 + i * 16 + mr;
        if (mA >= M) mA = M - 1;
        int batchA = mA / rpb;
        int rA     = mA - batchA * rpb;
        Arow[i] = A + (long long)batchA * aBatch + (long long)rA * K;
    }
    // B row pointers for the 4 N-frags
    const ushort* Brow[4];
#pragma unroll
    for (int j = 0; j < 4; ++j) {
        int nB = ng * 64 + j * 16 + mr;
        if (nB >= N) nB = N - 1;
        Brow[j] = Wt + (long long)nB * K;
    }

    v8f c[4][4] = {};
    for (int kk = 0; kk < K; kk += 32) {
        FragBF a[4], b[4];
#pragma unroll
        for (int i = 0; i < 4; ++i) {
            // A frag: VGPR v holds K pairs (group*16 + half*8 + (v&3)*2)
            a[i].q[0] = *(const uint4*)(Arow[i] + kk + hl * 8);
            a[i].q[1] = *(const uint4*)(Arow[i] + kk + 16 + hl * 8);
        }
#pragma unroll
        for (int j = 0; j < 4; ++j) {
            // B frag: lane's column n, same K order as A (group*16 + half*8 + (v&3)*2)
            b[j].q[0] = *(const uint4*)(Brow[j] + kk + hl * 8);
            b[j].q[1] = *(const uint4*)(Brow[j] + kk + 16 + hl * 8);
        }
#pragma unroll
        for (int i = 0; i < 4; ++i)
#pragma unroll
            for (int j = 0; j < 4; ++j)
                c[i][j] = __builtin_amdgcn_wmma_f32_16x16x32_bf16(
                    false, a[i].v, false, b[j].v, (short)0, c[i][j], false, false);
    }

#pragma unroll
    for (int j = 0; j < 4; ++j) {
        const int n = ng * 64 + j * 16 + (lane & 15);
        if (n >= N) continue;
        const float bv = (flags & 1) ? bias[n] : 0.f;
#pragma unroll
        for (int i = 0; i < 4; ++i) {
#pragma unroll
            for (int v = 0; v < 8; ++v) {
                int m = mg * 64 + i * 16 + v + hl * 8;  // C layout: M = v + 8*half
                if (m >= M) continue;
                int batch = m / rpb;
                int r     = m - batch * rpb;
                long long cidx = (long long)batch * cBatch + (long long)r * N + n;
                float val = c[i][j][v] + bv;
                if (flags & 4) val += resid[cidx];
                if (flags & 2) {
                    float x3 = val * val * val;
                    val = 0.5f * val * (1.f + tanhf(0.7978845608f * (val + 0.044715f * x3)));
                }
                if (flags & 8) Cb[cidx] = f2bf(val);
                else           Cf[cidx] = val;
            }
        }
    }
}

// ---------------- attention: one wave32 per (b,h,q) ------------------------
// qkv layout per row (b*T+t): [q(768) | k(768) | v(768)], head h at offset h*64
__global__ __launch_bounds__(256) void k_attn(const float* __restrict__ qkv,
                                              ushort* __restrict__ ctx, int T) {
    __shared__ float s_sc[8][S_];
    __shared__ float s_q[8][DH_];
    const int lane = threadIdx.x & 31;
    const int w    = threadIdx.x >> 5;
    long idx = (long)blockIdx.x * 8 + w;
    const long total = (long)B_ * H_ * T;
    if (idx >= total) idx = total - 1;            // clamp (no early return: barriers)
    const int b = (int)(idx / ((long)H_ * T));
    const int rem = (int)(idx - (long)b * H_ * T);
    const int h = rem / T;
    const int q = rem - h * T;

    const float* qrow = qkv + ((long)(b * T + q)) * (3 * D_) + h * DH_;
    s_q[w][lane * 2]     = qrow[lane * 2];
    s_q[w][lane * 2 + 1] = qrow[lane * 2 + 1];
    __syncthreads();

    float lmax = -1e30f;
    for (int k = lane; k <= q; k += 32) {
        const float* krow = qkv + ((long)(b * T + k)) * (3 * D_) + D_ + h * DH_;
        float s = 0.f;
        #pragma unroll 8
        for (int d = 0; d < DH_; ++d) s += s_q[w][d] * krow[d];
        s *= SCALE_;
        s_sc[w][k] = s;
        lmax = fmaxf(lmax, s);
    }
    for (int o = 16; o >= 1; o >>= 1) lmax = fmaxf(lmax, __shfl_xor(lmax, o, 32));
    __syncthreads();

    float lsum = 0.f;
    for (int k = lane; k <= q; k += 32) {
        float e = __expf(s_sc[w][k] - lmax);
        s_sc[w][k] = e;
        lsum += e;
    }
    for (int o = 16; o >= 1; o >>= 1) lsum += __shfl_xor(lsum, o, 32);
    __syncthreads();

    const float inv = 1.f / lsum;
    const int d0 = lane * 2;
    float a0 = 0.f, a1 = 0.f;
    for (int k = 0; k <= q; ++k) {
        const float p = s_sc[w][k];
        const float* vrow = qkv + ((long)(b * T + k)) * (3 * D_) + 2 * D_ + h * DH_;
        a0 += p * vrow[d0];
        a1 += p * vrow[d0 + 1];
    }
    ushort* crow = ctx + ((long)(b * T + q)) * D_ + h * DH_;
    crow[d0]     = f2bf(a0 * inv);
    crow[d0 + 1] = f2bf(a1 * inv);
}

// ---------------- latent feedback scatter ----------------------------------
__global__ __launch_bounds__(256) void k_latent_copy(const float* __restrict__ hid,
                                                     float* __restrict__ embeds,
                                                     int T, int slot) {
    int i = blockIdx.x * blockDim.x + threadIdx.x;   // over B*D
    if (i >= B_ * D_) return;
    int b = i / D_, d = i - (i / D_) * D_;
    embeds[((long)b * S_ + slot) * D_ + d] = hid[((long)b * T + (T - 1)) * D_ + d];
}

// ---------------- cross-entropy -------------------------------------------
// one block per row (b,t), t in [0, S-1); nll = lse(logits[b,t,:]) - logit[label[b,t+1]]
__global__ __launch_bounds__(256) void k_loss_rows(const float* __restrict__ logits,
                                                   const int* __restrict__ labels,
                                                   float* __restrict__ nll) {
    __shared__ float red[256];
    const long row = blockIdx.x;                 // 0 .. B*(S-1)-1
    const int b = (int)(row / (S_ - 1));
    const int t = (int)(row - (long)b * (S_ - 1));
    const float* lr = logits + ((long)b * S_ + t) * V_;
    const int tid = threadIdx.x;

    float m = -1e30f;
    for (int i = tid; i < V_; i += 256) m = fmaxf(m, lr[i]);
    red[tid] = m; __syncthreads();
    for (int s = 128; s >= 1; s >>= 1) {
        if (tid < s) red[tid] = fmaxf(red[tid], red[tid + s]);
        __syncthreads();
    }
    const float mx = red[0]; __syncthreads();

    float sum = 0.f;
    for (int i = tid; i < V_; i += 256) sum += __expf(lr[i] - mx);
    red[tid] = sum; __syncthreads();
    for (int s = 128; s >= 1; s >>= 1) {
        if (tid < s) red[tid] += red[tid + s];
        __syncthreads();
    }
    if (tid == 0) {
        int lab = labels[(long)b * S_ + t + 1];
        nll[row] = mx + logf(red[0]) - lr[lab];
    }
}

__global__ __launch_bounds__(256) void k_loss_reduce(const float* __restrict__ nll,
                                                     float* __restrict__ out, int n) {
    __shared__ float red[256];
    float s = 0.f;
    for (int i = threadIdx.x; i < n; i += 256) s += nll[i];
    red[threadIdx.x] = s; __syncthreads();
    for (int st = 128; st >= 1; st >>= 1) {
        if (threadIdx.x < st) red[threadIdx.x] += red[threadIdx.x + st];
        __syncthreads();
    }
    if (threadIdx.x == 0) out[0] = red[0] / (float)n;
}

// ---------------- host orchestration ---------------------------------------
static inline int cdiv_l(long a, long b) { return (int)((a + b - 1) / b); }

extern "C" void kernel_launch(void* const* d_in, const int* in_sizes, int n_in,
                              void* d_out, int out_size, void* d_ws, size_t ws_size,
                              hipStream_t stream) {
    (void)in_sizes; (void)n_in; (void)out_size; (void)ws_size;
    const int*   input_ids = (const int*)  d_in[0];
    const int*   labels    = (const int*)  d_in[2];
    const float* wte  = (const float*)d_in[4];
    const float* wpe  = (const float*)d_in[5];
    const float* ln1g = (const float*)d_in[6];
    const float* ln1b = (const float*)d_in[7];
    const float* Wqkv = (const float*)d_in[8];
    const float* bqkv = (const float*)d_in[9];
    const float* Wo   = (const float*)d_in[10];
    const float* bo   = (const float*)d_in[11];
    const float* ln2g = (const float*)d_in[12];
    const float* ln2b = (const float*)d_in[13];
    const float* W1   = (const float*)d_in[14];
    const float* b1   = (const float*)d_in[15];
    const float* W2   = (const float*)d_in[16];
    const float* b2   = (const float*)d_in[17];
    const float* lnfg = (const float*)d_in[18];
    const float* lnfb = (const float*)d_in[19];
    float* out = (float*)d_out;            // out[0]=loss, out[1..]=logits [B,S,V]

    // ---- workspace bump allocator
    char* p = (char*)d_ws;
    auto alloc = [&](size_t bytes) -> char* {
        char* r = p;
        p += (bytes + 255) & ~(size_t)255;
        return r;
    };
    ushort* wteb   = (ushort*)alloc((size_t)V_ * D_ * 2);
    ushort* WqkvT  = (ushort*)alloc((size_t)NL_ * 3 * D_ * D_ * 2);
    ushort* WoT    = (ushort*)alloc((size_t)NL_ * D_ * D_ * 2);
    ushort* W1T    = (ushort*)alloc((size_t)NL_ * D_ * DFF_ * 2);
    ushort* W2T    = (ushort*)alloc((size_t)NL_ * DFF_ * D_ * 2);
    float*  embeds = (float*) alloc((size_t)B_ * S_ * D_ * 4);
    float*  x      = (float*) alloc((size_t)B_ * S_ * D_ * 4);
    ushort* h_bf   = (ushort*)alloc((size_t)B_ * S_ * D_ * 2);
    float*  qkv    = (float*) alloc((size_t)B_ * S_ * 3 * D_ * 4);
    ushort* ctx_bf = (ushort*)alloc((size_t)B_ * S_ * D_ * 2);
    ushort* ff_bf  = (ushort*)alloc((size_t)B_ * S_ * DFF_ * 2);
    float*  hid_f  = (float*) alloc((size_t)B_ * S_ * D_ * 4);
    ushort* hid_bf = (ushort*)alloc((size_t)B_ * S_ * D_ * 2);
    float*  nll    = (float*) alloc((size_t)B_ * (S_ - 1) * 4);

    // ---- one-time (per launch) weight conversion to bf16, transposed [N,K]
    k_convert<<<cdiv_l((long)V_ * D_, 256), 256, 0, stream>>>(wte, wteb, (long)V_ * D_);
    k_convert_t<<<cdiv_l((long)NL_ * D_ * 3 * D_, 256), 256, 0, stream>>>(Wqkv, WqkvT, NL_, D_, 3 * D_);
    k_convert_t<<<cdiv_l((long)NL_ * D_ * D_, 256), 256, 0, stream>>>(Wo, WoT, NL_, D_, D_);
    k_convert_t<<<cdiv_l((long)NL_ * D_ * DFF_, 256), 256, 0, stream>>>(W1, W1T, NL_, D_, DFF_);
    k_convert_t<<<cdiv_l((long)NL_ * DFF_ * D_, 256), 256, 0, stream>>>(W2, W2T, NL_, DFF_, D_);
    k_embed<<<cdiv_l((long)B_ * S_ * D_, 256), 256, 0, stream>>>(input_ids, wte, embeds);

    auto gemm_blocks = [](int M, int N) -> int {
        long tiles = (long)((M + 63) / 64) * ((N + 63) / 64);
        return (int)((tiles + 7) / 8);
    };

    const int Ts[7] = {64, 65, 66, 67, 68, 69, 512};
    const int As[7] = {0, 64, 65, 66, 67, 68, 69};

    for (int ps = 0; ps < N_LATENT_ + 1; ++ps) {
        const int T = Ts[ps];
        const int a = As[ps];
        const int M = B_ * T;

        k_xinit<<<cdiv_l((long)M * D_, 256), 256, 0, stream>>>(embeds, wpe, x, T);

        for (int l = 0; l < NL_; ++l) {
            // ---- attention block
            k_ln<<<cdiv_l(M, 8), 256, 0, stream>>>(x, ln1g + l * D_, ln1b + l * D_,
                                                   h_bf, nullptr, M);
            k_gemm<<<gemm_blocks(M, 3 * D_), 256, 0, stream>>>(
                h_bf, WqkvT + (size_t)l * 3 * D_ * D_, bqkv + (size_t)l * 3 * D_,
                nullptr, qkv, nullptr, M, 3 * D_, D_, M, 0LL, 0LL, /*bias*/1);
            k_attn<<<cdiv_l((long)B_ * H_ * T, 8), 256, 0, stream>>>(qkv, ctx_bf, T);
            k_gemm<<<gemm_blocks(M, D_), 256, 0, stream>>>(
                ctx_bf, WoT + (size_t)l * D_ * D_, bo + (size_t)l * D_,
                x, x, nullptr, M, D_, D_, M, 0LL, 0LL, /*bias|resid*/1 | 4);
            // ---- MLP block
            k_ln<<<cdiv_l(M, 8), 256, 0, stream>>>(x, ln2g + l * D_, ln2b + l * D_,
                                                   h_bf, nullptr, M);
            k_gemm<<<gemm_blocks(M, DFF_), 256, 0, stream>>>(
                h_bf, W1T + (size_t)l * DFF_ * D_, b1 + (size_t)l * DFF_,
                nullptr, nullptr, ff_bf, M, DFF_, D_, M, 0LL, 0LL, /*bias|gelu|bf16out*/1 | 2 | 8);
            k_gemm<<<gemm_blocks(M, D_), 256, 0, stream>>>(
                ff_bf, W2T + (size_t)l * D_ * DFF_, b2 + (size_t)l * D_,
                x, x, nullptr, M, D_, DFF_, M, 0LL, 0LL, /*bias|resid*/1 | 4);
        }

        // ---- final LN -> hid (f32 for latent feedback, bf16 for lm head)
        k_ln<<<cdiv_l(M, 8), 256, 0, stream>>>(x, lnfg, lnfb, hid_bf, hid_f, M);

        // ---- lm head: logits for positions [a, T)  ->  out[1 + (b*S + t)*V + n]
        const int RT = T - a;
        const int Mq = B_ * RT;
        k_gemm<<<gemm_blocks(Mq, V_), 256, 0, stream>>>(
            hid_bf + (size_t)a * D_, wteb, nullptr, nullptr,
            out + 1 + (long long)a * V_, nullptr,
            Mq, V_, D_, RT, (long long)T * D_, (long long)S_ * V_, /*plain f32*/0);

        if (ps < N_LATENT_) {
            k_latent_copy<<<cdiv_l((long)B_ * D_, 256), 256, 0, stream>>>(
                hid_f, embeds, T, LATENT_START_ + ps);
        }
    }

    // ---- cross-entropy over shifted logits/labels
    k_loss_rows<<<B_ * (S_ - 1), 256, 0, stream>>>(out + 1, labels, nll);
    k_loss_reduce<<<1, 256, 0, stream>>>(nll, out, B_ * (S_ - 1));
}
